// NeighbourApproxPCA_70781061038811
// MI455X (gfx1250) — matrix operand order repaired and checked by reading, not compiled, not run on hardware
//
#include <hip/hip_runtime.h>
#include <hip/hip_bf16.h>

// NeighbourApproxPCA fused kernel for gfx1250 (MI455X, wave32, WMMA).
// V vertices, K=64 neighbours, C=3 coords, F=16 features.
// Stage 1: per-(vertex,feature) weighted 3x3 covariance (half-wave per vertex,
//          lane = feature).
// Stage 2: 4-layer MLP 9->64->64->64->9 with ELU, via v_wmma_f32_16x16x32_f16.
//          One vertex's 16 features == one 16-row WMMA tile.

typedef __attribute__((ext_vector_type(16))) _Float16 v16h;
typedef __attribute__((ext_vector_type(8)))  _Float16 h8;
typedef __attribute__((ext_vector_type(8)))  float    v8f;

union Frag { v16h v; h8 h[2]; };

__device__ __forceinline__ float elu_f(float x) {
    return x > 0.0f ? x : (__expf(x) - 1.0f);
}

// A-fragment (16x32 f16) from LDS tile X (row stride 80 halves, 16B aligned).
// ISA layout: lanes 0-15: M=lane, elems 0-7 = K0..7, elems 8-15 = K16..23
//             lanes 16-31: M=lane-16, elems 0-7 = K8..15, elems 8-15 = K24..31
__device__ __forceinline__ v16h load_a(const _Float16* Xw, int kc, int lane) {
    int m  = lane & 15;
    int kb = kc * 32 + ((lane & 16) ? 8 : 0);
    Frag a;
    a.h[0] = *(const h8*)&Xw[m * 80 + kb];
    a.h[1] = *(const h8*)&Xw[m * 80 + kb + 16];
    return a.v;
}

// B-fragment (32x16 f16) from LDS weights stored transposed WT[N][Kpad].
// ISA layout: lanes 0-15: N=lane, elems = K0..7,K16..23 ; lanes 16-31: N=lane-16,
//             elems = K8..15,K24..31 (offset by kc*32 for the second K chunk).
__device__ __forceinline__ v16h load_b(const _Float16* WT, int stride, int nt,
                                       int kc, int lane) {
    int n  = nt * 16 + (lane & 15);
    int kb = kc * 32 + ((lane & 16) ? 8 : 0);
    Frag b;
    b.h[0] = *(const h8*)&WT[n * stride + kb];
    b.h[1] = *(const h8*)&WT[n * stride + kb + 16];
    return b.v;
}

__global__ __launch_bounds__(128) void pca_fused_kernel(
    const float* __restrict__ coords, const float* __restrict__ distsq,
    const float* __restrict__ feats,  const int*   __restrict__ nidx,
    const float* __restrict__ W0, const float* __restrict__ b0,
    const float* __restrict__ W1, const float* __restrict__ b1,
    const float* __restrict__ W2, const float* __restrict__ b2,
    const float* __restrict__ W3, const float* __restrict__ b3,
    float* __restrict__ out, int V)
{
    // f16 weights, transposed to [N][Kpad] so B-frag loads are contiguous.
    __shared__ __align__(16) _Float16 WT0[64 * 32];   // K 9->32 zero padded
    __shared__ __align__(16) _Float16 WT1[64 * 64];
    __shared__ __align__(16) _Float16 WT2[64 * 64];
    __shared__ __align__(16) _Float16 WT3[16 * 64];   // N 9->16 zero padded
    __shared__ float BS[208];                          // b0|b1|b2|b3(pad)
    __shared__ __align__(16) _Float16 X[4][16 * 80];   // per-wave activation tile

    const int tid = threadIdx.x;

    for (int i = tid; i < 64 * 32; i += 128) {
        int n = i >> 5, kk = i & 31;
        WT0[i] = (_Float16)((kk < 9) ? W0[kk * 64 + n] : 0.0f);
    }
    for (int i = tid; i < 64 * 64; i += 128) {
        int n = i >> 6, kk = i & 63;
        WT1[i] = (_Float16)W1[kk * 64 + n];
        WT2[i] = (_Float16)W2[kk * 64 + n];
    }
    for (int i = tid; i < 16 * 64; i += 128) {
        int n = i >> 6, kk = i & 63;
        WT3[i] = (_Float16)((n < 9) ? W3[kk * 9 + n] : 0.0f);
    }
    for (int i = tid; i < 208; i += 128) {
        float bv;
        if      (i < 64)  bv = b0[i];
        else if (i < 128) bv = b1[i - 64];
        else if (i < 192) bv = b2[i - 128];
        else              bv = (i - 192 < 9) ? b3[i - 192] : 0.0f;
        BS[i] = bv;
    }
    __syncthreads();

    const int lane  = tid & 31;
    const int wid   = tid >> 5;
    const int f     = lane & 15;      // feature index
    const int hw    = lane >> 4;      // half-wave id (vertex within wave)
    const int vbase = blockIdx.x * 8 + wid * 2;
    const int v     = vbase + hw;     // this half-wave's vertex (cov stage)

    // prefetch next block's streamed rows into L2 (global_prefetch_b8)
    if (vbase + 8 < V) {
        __builtin_prefetch(&nidx[(size_t)(vbase + 8) * 64], 0, 1);
        __builtin_prefetch(&distsq[(size_t)(vbase + 8) * 64], 0, 1);
    }

    // ---------------- Stage 1: weighted neighbour covariance ----------------
    float wsum = 0.f, sx = 0.f, sy = 0.f, sz = 0.f;
    float sxx = 0.f, sxy = 0.f, sxz = 0.f, syy = 0.f, syz = 0.f, szz = 0.f;

    int4   i4 = {0, 0, 0, 0};
    float4 e4 = {0.f, 0.f, 0.f, 0.f};
    if (v < V) {
        i4 = *(const int4*)&nidx[(size_t)v * 64 + f * 4];
        float4 d4 = *(const float4*)&distsq[(size_t)v * 64 + f * 4];
        e4.x = __expf(-10.0f * d4.x);
        e4.y = __expf(-10.0f * d4.y);
        e4.z = __expf(-10.0f * d4.z);
        e4.w = __expf(-10.0f * d4.w);
    }

    for (int kq = 0; kq < 16; ++kq) {
        int src = (hw << 4) + kq;     // broadcast source lane in this half-wave
        int   ii[4];
        float ee[4];
        ii[0] = __shfl(i4.x, src, 32); ee[0] = __shfl(e4.x, src, 32);
        ii[1] = __shfl(i4.y, src, 32); ee[1] = __shfl(e4.y, src, 32);
        ii[2] = __shfl(i4.z, src, 32); ee[2] = __shfl(e4.z, src, 32);
        ii[3] = __shfl(i4.w, src, 32); ee[3] = __shfl(e4.w, src, 32);
        #pragma unroll
        for (int c4 = 0; c4 < 4; ++c4) {
            int   idx = ii[c4];
            float w   = feats[(size_t)idx * 16 + f] * ee[c4];  // coalesced 64B
            const float* cp = &coords[(size_t)idx * 3];        // L2-resident
            float cx = cp[0], cy = cp[1], cz = cp[2];
            wsum += w;
            sx += w * cx;  sy += w * cy;  sz += w * cz;
            sxx += w * cx * cx; sxy += w * cx * cy; sxz += w * cx * cz;
            syy += w * cy * cy; syz += w * cy * cz; szz += w * cz * cz;
        }
    }

    const float inv = 1.0f / (wsum + 1e-4f);
    const float mx = sx * inv, my = sy * inv, mz = sz * inv;
    float cov9[9];
    cov9[0] = sxx * inv - mx * mx;  cov9[1] = sxy * inv - mx * my;
    cov9[2] = sxz * inv - mx * mz;  cov9[3] = sxy * inv - my * mx;
    cov9[4] = syy * inv - my * my;  cov9[5] = syz * inv - my * mz;
    cov9[6] = sxz * inv - mz * mx;  cov9[7] = syz * inv - mz * my;
    cov9[8] = szz * inv - mz * mz;

    // ---------------- Stage 2: WMMA MLP, one vertex = one 16-row tile -------
    _Float16* Xw = X[wid];

    for (int sub = 0; sub < 2; ++sub) {
        int vv = vbase + sub;
        if (vv >= V) break;           // wave-uniform

        // fill layer-0 input rows (cols 0..8 cov, 9..31 zero) from owning half
        if (hw == sub) {
            #pragma unroll
            for (int c = 0; c < 9; ++c)  Xw[f * 80 + c] = (_Float16)cov9[c];
            #pragma unroll
            for (int c = 9; c < 32; ++c) Xw[f * 80 + c] = (_Float16)0.0f;
        }
        // same-wave LDS ops are in-order: stores above are seen by loads below

        const int nc = lane & 15;
        const int rb = (lane & 16) ? 8 : 0;

        // ---- layer 0: [16,32pad] x [32,64] ----
        {
            v16h a = load_a(Xw, 0, lane);
            #pragma unroll
            for (int nt = 0; nt < 4; ++nt) {
                float bv = BS[nt * 16 + nc];
                v8f c = {bv, bv, bv, bv, bv, bv, bv, bv};
                v16h bfr = load_b(WT0, 32, nt, 0, lane);
                c = __builtin_amdgcn_wmma_f32_16x16x32_f16(
                        false, a, false, bfr, (short)0, c, false, false);
                #pragma unroll
                for (int j = 0; j < 8; ++j)
                    Xw[(j + rb) * 80 + nt * 16 + nc] = (_Float16)elu_f(c[j]);
            }
        }

        // ---- layers 1 & 2: [16,64] x [64,64] ----
        #pragma unroll
        for (int l = 0; l < 2; ++l) {
            const _Float16* WT = (l == 0) ? WT1 : WT2;
            const int boff = 64 + l * 64;
            v16h a0 = load_a(Xw, 0, lane);   // load full input before stores
            v16h a1 = load_a(Xw, 1, lane);
            #pragma unroll
            for (int nt = 0; nt < 4; ++nt) {
                float bv = BS[boff + nt * 16 + nc];
                v8f c = {bv, bv, bv, bv, bv, bv, bv, bv};
                v16h bf0 = load_b(WT, 64, nt, 0, lane);
                c = __builtin_amdgcn_wmma_f32_16x16x32_f16(
                        false, a0, false, bf0, (short)0, c, false, false);
                v16h bf1 = load_b(WT, 64, nt, 1, lane);
                c = __builtin_amdgcn_wmma_f32_16x16x32_f16(
                        false, a1, false, bf1, (short)0, c, false, false);
                #pragma unroll
                for (int j = 0; j < 8; ++j)
                    Xw[(j + rb) * 80 + nt * 16 + nc] = (_Float16)elu_f(c[j]);
            }
        }

        // ---- layer 3: [16,64] x [64,16pad], ELU, scatter 9 cols to global --
        {
            v16h a0 = load_a(Xw, 0, lane);
            v16h a1 = load_a(Xw, 1, lane);
            float bv = BS[192 + nc];
            v8f c = {bv, bv, bv, bv, bv, bv, bv, bv};
            v16h bf0 = load_b(WT3, 64, 0, 0, lane);
            c = __builtin_amdgcn_wmma_f32_16x16x32_f16(
                    false, a0, false, bf0, (short)0, c, false, false);
            v16h bf1 = load_b(WT3, 64, 0, 1, lane);
            c = __builtin_amdgcn_wmma_f32_16x16x32_f16(
                    false, a1, false, bf1, (short)0, c, false, false);
            if (nc < 9) {
                float* op = out + (size_t)vv * 144 + nc;
                #pragma unroll
                for (int j = 0; j < 8; ++j)
                    op[(j + rb) * 9] = elu_f(c[j]);
            }
        }
    }
}

extern "C" void kernel_launch(void* const* d_in, const int* in_sizes, int n_in,
                              void* d_out, int out_size, void* d_ws, size_t ws_size,
                              hipStream_t stream) {
    const float* coords = (const float*)d_in[0];
    const float* distsq = (const float*)d_in[1];
    const float* feats  = (const float*)d_in[2];
    const int*   nidx   = (const int*)  d_in[3];
    const float* W0 = (const float*)d_in[4];  const float* b0 = (const float*)d_in[5];
    const float* W1 = (const float*)d_in[6];  const float* b1 = (const float*)d_in[7];
    const float* W2 = (const float*)d_in[8];  const float* b2 = (const float*)d_in[9];
    const float* W3 = (const float*)d_in[10]; const float* b3 = (const float*)d_in[11];
    float* out = (float*)d_out;

    const int V = in_sizes[0] / 3;            // coordinates is [V,3]
    const int blocks = (V + 7) / 8;           // 4 waves/block, 2 vertices/wave

    pca_fused_kernel<<<dim3(blocks), dim3(128), 0, stream>>>(
        coords, distsq, feats, nidx, W0, b0, W1, b1, W2, b2, W3, b3, out, V);
}
